// NeighborsValuesAssigner_8529805050216
// MI455X (gfx1250) — matrix operand order repaired and checked by reading, not compiled, not run on hardware
//
#include <hip/hip_runtime.h>
#include <hip/hip_bf16.h>

typedef __attribute__((ext_vector_type(16))) _Float16 v16h;
typedef __attribute__((ext_vector_type(8)))  _Float16 v8h;
typedef __attribute__((ext_vector_type(8)))  float    v8f;

#define NPATCH 2048
#define KREAL  75
#define KPAD   96
#define DROW   132      // padded LDS row stride (floats) for the distance tile
#define NCHUNK 128
#define NCHUNKS (NPATCH / NCHUNK)   // 16

// ---------------------------------------------------------------------------
// Prep: Bmat[n][k] = (fp16)(-patches[n][k]) (K padded 75->96 with zeros),
//       bias[n]   = 0.5 * ||patch_n||^2  (fp32)
// Row-major K per patch => a WMMA B-fragment is one contiguous 32B load/lane.
// ---------------------------------------------------------------------------
__global__ void nva_prep(const float* __restrict__ patches,
                         _Float16* __restrict__ Bmat,
                         float* __restrict__ bias) {
    __shared__ float sq[KPAD];
    const int n = blockIdx.x;
    const int k = threadIdx.x;            // 96 threads
    float v = 0.0f;
    if (k < KREAL) v = patches[n * KREAL + k];
    Bmat[n * KPAD + k] = (_Float16)(-v);
    sq[k] = v * v;
    __syncthreads();
    if (k == 0) {
        float s = 0.0f;
        for (int i = 0; i < KREAL; ++i) s += sq[i];
        bias[n] = 0.5f * s;
    }
}

// streaming top-8 (smallest) insertion with tracked worst slot
__device__ __forceinline__ void top8_insert(float v, int n,
                                            float* bv, int* bi,
                                            float& wval, int& wslot) {
    if (v < wval) {
        bv[wslot] = v; bi[wslot] = n;
        wval = bv[0]; wslot = 0;
        #pragma unroll
        for (int j = 1; j < 8; ++j)
            if (bv[j] > wval) { wval = bv[j]; wslot = j; }
    }
}

// ---------------------------------------------------------------------------
// Fused conv-as-GEMM (WMMA) + streaming top-8 + gather/mean.
// One block = one (b, h) image row = 64 pixels (w = 0..63).
// 8 waves: wave w -> m_tile = w>>1 (16 pixels), n-half = w&1 (4 n-tiles/chunk).
// ---------------------------------------------------------------------------
__global__ __launch_bounds__(256) void nva_fused(
        const float* __restrict__ x,        // (32,3,64,64)
        const _Float16* __restrict__ Bmat,  // (2048,96) fp16, negated patches
        const float* __restrict__ bias,     // (2048)
        const float* __restrict__ values,   // (2048,128)
        float* __restrict__ out) {          // (32,128,64,64)

    __shared__ _Float16 Ash[64 * KPAD];     // 12 KB im2col tile (fp16)
    __shared__ float    Dsh[64 * DROW];     // 33 KB distance tile / scratch
    __shared__ float    biasSh[NPATCH];     //  8 KB staged bias

    const int t = threadIdx.x;
    const int b = blockIdx.x >> 6;          // batch
    const int h = blockIdx.x & 63;          // image row

    // ---- stage bias in LDS (read 64x per lane later) ----
    #pragma unroll
    for (int i = t; i < NPATCH; i += 256) biasSh[i] = bias[i];

    // ---- im2col: A[m][k], m == w, zero 'same' padding, K padded to 96 ----
    for (int i = t; i < 64 * KPAD; i += 256) {
        const int m = i / KPAD;
        const int k = i - m * KPAD;
        float v = 0.0f;
        if (k < KREAL) {
            const int c  = k / 25;
            const int r  = k - c * 25;
            const int kh = r / 5;
            const int kw = r - kh * 5;
            const int ih = h + kh - 2;
            const int iw = m + kw - 2;
            if (ih >= 0 && ih < 64 && iw >= 0 && iw < 64)
                v = x[((b * 3 + c) * 64 + ih) * 64 + iw];
        }
        Ash[i] = (_Float16)v;
    }
    __syncthreads();

    const int wv     = t >> 5;              // wave 0..7
    const int l      = t & 31;              // lane (wave32)
    const int m_tile = wv >> 1;             // 0..3
    const int nhalf  = wv & 1;              // 0..1
    const int lrow   = l & 15;
    const int aLo    = (l < 16) ? 0 : 16;   // 16-bit A-fragment K sub-offset
    const int bLo    = (l < 16) ? 0 : 16;   // 16-bit B-fragment K sub-offset

    // ---- load the 3 A-fragments (K = 0..31, 32..63, 64..95) from LDS ----
    // A 16x32 fp16 layout: halfs = K{lo..lo+15} (contiguous, matches B)
    v16h afr[3];
    #pragma unroll
    for (int s = 0; s < 3; ++s) {
        const _Float16* p = &Ash[(m_tile * 16 + lrow) * KPAD + s * 32 + aLo];
        const v8h p0 = *(const v8h*)(p);        // K base+lo + 0..7
        const v8h p1 = *(const v8h*)(p + 8);    // K base+lo + 8..15
        #pragma unroll
        for (int j = 0; j < 8; ++j) { afr[s][j] = p0[j]; afr[s][8 + j] = p1[j]; }
    }

    // ---- private streaming top-8 state: 4 threads per pixel ----
    float bv[8]; int bi[8];
    #pragma unroll
    for (int j = 0; j < 8; ++j) { bv[j] = 3.0e38f; bi[j] = 0; }
    float wval = 3.0e38f; int wslot = 0;     // current worst (max) slot
    const int pm   = t >> 2;                 // scanned pixel 0..63
    const int sub  = t & 3;                  // contiguous 32-col segment id
    const int scol = sub * 32;               // segment start column

    for (int chunk = 0; chunk < NCHUNKS; ++chunk) {
        const int nbase = chunk * NCHUNK;

        // prefetch next dictionary chunk into L2 (speculative)
        if (chunk + 1 < NCHUNKS)
            __builtin_prefetch((const char*)(Bmat + (size_t)(nbase + NCHUNK) * KPAD) + t * 96, 0, 1);

        // ---- 4 WMMA tiles per wave: D = A(16x96) * B(96x16) + bias ----
        #pragma unroll
        for (int j4 = 0; j4 < 4; ++j4) {
            const int ntl = nhalf * 4 + j4;              // local n-tile 0..7
            const int n   = nbase + ntl * 16 + lrow;     // this lane's column
            const float bsv = biasSh[n];
            v8f acc;
            #pragma unroll
            for (int i = 0; i < 8; ++i) acc[i] = bsv;    // fold +bias into C init
            const _Float16* bp = Bmat + (size_t)n * KPAD + bLo;
            #pragma unroll
            for (int s = 0; s < 3; ++s) {
                const v16h bfr = *(const v16h*)(bp + s * 32);   // contiguous 32B
                acc = __builtin_amdgcn_wmma_f32_16x16x32_f16(
                        false, afr[s], false, bfr, (short)0, acc, false, false);
            }
            // scatter D per C/D layout: VGPR r -> row r (+8 for hi lanes)
            const int row0 = m_tile * 16 + ((l < 16) ? 0 : 8);
            const int col  = ntl * 16 + lrow;
            #pragma unroll
            for (int r = 0; r < 8; ++r) Dsh[(row0 + r) * DROW + col] = acc[r];
        }
        __syncthreads();

        // ---- streaming top-8 update: contiguous segment, float4 LDS reads ----
        const float* drow = &Dsh[pm * DROW + scol];
        #pragma unroll
        for (int c4 = 0; c4 < 8; ++c4) {
            const float4 v4 = *(const float4*)(drow + c4 * 4);
            const int nb = nbase + scol + c4 * 4;
            top8_insert(v4.x, nb + 0, bv, bi, wval, wslot);
            top8_insert(v4.y, nb + 1, bv, bi, wval, wslot);
            top8_insert(v4.z, nb + 2, bv, bi, wval, wslot);
            top8_insert(v4.w, nb + 3, bv, bi, wval, wslot);
        }
        __syncthreads();   // Dsh reused next chunk
    }

    // ---- merge 4 partial top-8 lists per pixel (reuse Dsh as scratch) ----
    float* candV = Dsh;                      // [64][4][8] floats
    int*   candI = (int*)(Dsh + 2048);       // [64][4][8] ints
    int*   idxF  = (int*)(Dsh + 4096);       // [64][8] final indices
    #pragma unroll
    for (int j = 0; j < 8; ++j) {
        candV[(pm * 4 + sub) * 8 + j] = bv[j];
        candI[(pm * 4 + sub) * 8 + j] = bi[j];
    }
    __syncthreads();
    if (t < 64) {
        float fv[8]; int fi[8];
        #pragma unroll
        for (int j = 0; j < 8; ++j) { fv[j] = 3.0e38f; fi[j] = 0; }
        float wv2 = 3.0e38f; int ws2 = 0;
        for (int c = 0; c < 32; ++c) {
            const float v = candV[t * 32 + c];
            if (v < wv2) {
                fv[ws2] = v; fi[ws2] = candI[t * 32 + c];
                wv2 = fv[0]; ws2 = 0;
                #pragma unroll
                for (int j = 1; j < 8; ++j)
                    if (fv[j] > wv2) { wv2 = fv[j]; ws2 = j; }
            }
        }
        #pragma unroll
        for (int j = 0; j < 8; ++j) idxF[t * 8 + j] = fi[j];
    }
    __syncthreads();

    // ---- gather values[idx] and mean over K=8; coalesced over w ----
    const int gm = t & 63;                   // pixel (== w), contiguous per lane
    const int gq = t >> 6;                   // quarter of the 128 dims
    int idxs[8];
    #pragma unroll
    for (int j = 0; j < 8; ++j) idxs[j] = idxF[gm * 8 + j];
    for (int dd = 0; dd < 32; dd += 4) {
        const int d = gq * 32 + dd;
        float4 a = make_float4(0.f, 0.f, 0.f, 0.f);
        #pragma unroll
        for (int j = 0; j < 8; ++j) {
            const float4 vv = *(const float4*)(values + (size_t)idxs[j] * 128 + d);
            a.x += vv.x; a.y += vv.y; a.z += vv.z; a.w += vv.w;
        }
        const size_t obase = (((size_t)b * 128 + d) * 64 + h) * 64 + gm;
        out[obase]            = a.x * 0.125f;
        out[obase + 4096]     = a.y * 0.125f;
        out[obase + 2 * 4096] = a.z * 0.125f;
        out[obase + 3 * 4096] = a.w * 0.125f;
    }
}

// ---------------------------------------------------------------------------
extern "C" void kernel_launch(void* const* d_in, const int* in_sizes, int n_in,
                              void* d_out, int out_size, void* d_ws, size_t ws_size,
                              hipStream_t stream) {
    const float* x       = (const float*)d_in[0];   // (32,3,64,64)
    const float* patches = (const float*)d_in[1];   // (2048,3,5,5)
    const float* values  = (const float*)d_in[2];   // (2048,128)
    float* out = (float*)d_out;                     // (32,128,64,64)

    // workspace: bias (2048 f32) then Bmat (2048 x 96 f16)
    float*    bias = (float*)d_ws;
    _Float16* Bmat = (_Float16*)((char*)d_ws + NPATCH * sizeof(float));

    nva_prep<<<NPATCH, KPAD, 0, stream>>>(patches, Bmat, bias);
    nva_fused<<<32 * 64, 256, 0, stream>>>(x, Bmat, bias, values, out);
}
